// ConditionedISFNOTransition_30305289240730
// MI455X (gfx1250) — matrix operand order repaired and checked by reading, not compiled, not run on hardware
//
#include <hip/hip_runtime.h>
#include <hip/hip_bf16.h>
#include <math.h>

// ---------------------------------------------------------------------------
// Types for CDNA5 WMMA (wave32): v_wmma_f32_16x16x32_bf16
// ---------------------------------------------------------------------------
typedef __attribute__((ext_vector_type(16))) __bf16 v16bf;
typedef __attribute__((ext_vector_type(8)))  __bf16 bf16x8;
typedef __attribute__((ext_vector_type(8)))  float  v8f;

#define BATCH 16384
#define DDIM  256
#define HID   1024
#define NOBS  20
#define UDIM  16

#define TILE_M 128
#define TILE_N 128
#define TILE_K 32
#define LPITCH 40  // bf16 elems per LDS row: 32 + 8 pad (80B, 16B aligned, conflict-free)

__device__ inline v8f vzero8() {
  v8f z = {0.f,0.f,0.f,0.f,0.f,0.f,0.f,0.f};
  return z;
}

// ---------------------------------------------------------------------------
// CDNA5 async memory->LDS copy (GLOBAL_LOAD_ASYNC_TO_LDS_B128, ASYNCcnt).
// LDS byte address = low 32 bits of the generic pointer to __shared__ memory.
// ---------------------------------------------------------------------------
__device__ inline void async_copy16(const __bf16* gsrc, __bf16* lds_dst) {
  const unsigned lds_addr = (unsigned)(uintptr_t)lds_dst;
  const unsigned long long gaddr = (unsigned long long)(uintptr_t)gsrc;
  asm volatile("global_load_async_to_lds_b128 %0, %1, off"
               :: "v"(lds_addr), "v"(gaddr)
               : "memory");
}

__device__ inline void wait_asynccnt0() {
  asm volatile("s_wait_asynccnt 0x0" ::: "memory");
}

// A-fragment (16x32 bf16, M x K). ISA 7.12.2: lane half hi -> elems[0..7]=K(8*hi..),
// elems[8..15]=K(16+8*hi..), row m = lane&15.
__device__ inline v16bf load_frag_A(const __bf16* lds, int row_base) {
  const int lane = threadIdx.x & 31;
  const int m = lane & 15, hi = lane >> 4;
  const __bf16* r = lds + (row_base + m) * LPITCH;
  union { v16bf v; bf16x8 h[2]; } u;
  u.h[0] = *(const bf16x8*)(r + 8 * hi);
  u.h[1] = *(const bf16x8*)(r + 16 + 8 * hi);
  return u.v;
}

// B-fragment (32x16 bf16, K x N) from W rows (W is [N,K] row-major, so B = W^T tile).
// lane half hi -> elems[0..7]=K(8*hi..), elems[8..15]=K(16+8*hi..); col n = lane&15.
__device__ inline v16bf load_frag_B(const __bf16* lds, int col_base) {
  const int lane = threadIdx.x & 31;
  const int n = lane & 15, hi = lane >> 4;
  const __bf16* r = lds + (col_base + n) * LPITCH + 8 * hi;
  union { v16bf v; bf16x8 h[2]; } u;
  u.h[0] = *(const bf16x8*)(r);
  u.h[1] = *(const bf16x8*)(r + 16);
  return u.v;
}

// ---------------------------------------------------------------------------
// Tiled bf16 WMMA GEMM: Y = epilogue(X @ W^T + bias)
//   X: [M,K] bf16, W: [N,K] bf16 row-major, bias: [N] f32 (nullable)
//   gelu!=0 : Y = gelu_exact(acc+bias)          -> Ybf
//   gelu==0 : v = alpha*(acc+bias) + beta*R     -> Ybf (if non-null), Yf (if non-null)
// 8 waves: 4x2 wave grid, each wave 32x64 = 2x4 WMMA tiles.
// Double-buffered LDS, tiles staged with async-to-LDS (ASYNCcnt), one barrier/step.
// ---------------------------------------------------------------------------
__global__ __launch_bounds__(256)
void gemm_bf16_wmma(const __bf16* __restrict__ X, const __bf16* __restrict__ W,
                    const float* __restrict__ bias, const float* __restrict__ R,
                    float alpha, float beta, int gelu,
                    __bf16* __restrict__ Ybf, float* __restrict__ Yf,
                    int M, int N, int K) {
  __shared__ __bf16 As[2][TILE_M * LPITCH];
  __shared__ __bf16 Bs[2][TILE_N * LPITCH];

  const int bm   = blockIdx.x * TILE_M;
  const int bn   = blockIdx.y * TILE_N;
  const int tid  = threadIdx.x;
  const int lane = tid & 31;
  const int wave = tid >> 5;
  const int wm   = (wave & 3) * 32;  // wave row offset in block tile
  const int wn   = (wave >> 2) * 64; // wave col offset in block tile

  v8f acc[2][4];
#pragma unroll
  for (int i = 0; i < 2; ++i)
#pragma unroll
    for (int j = 0; j < 4; ++j) acc[i][j] = vzero8();

  // Per-thread staging chunks: 128 rows x 32 bf16 = 512 x 16B chunks, 2 per thread
  // for A and 2 for B per K-tile.
  auto stage = [&](int buf, int k0) {
#pragma unroll
    for (int c = 0; c < 2; ++c) {
      const int chunk = tid + c * 256;
      const int row = chunk >> 2;
      const int kk  = (chunk & 3) * 8;
      async_copy16(X + (size_t)(bm + row) * K + k0 + kk, &As[buf][row * LPITCH + kk]);
      async_copy16(W + (size_t)(bn + row) * K + k0 + kk, &Bs[buf][row * LPITCH + kk]);
    }
  };

  const int ntiles = K / TILE_K;
  stage(0, 0);

  for (int t = 0; t < ntiles; ++t) {
    const int cur = t & 1;
    wait_asynccnt0();   // my async writes for tile t have landed in LDS
    __syncthreads();    // everyone's writes landed; everyone done reading buf cur^1
    if (t + 1 < ntiles) stage(cur ^ 1, (t + 1) * TILE_K);

    v16bf afr[2], bfr[4];
#pragma unroll
    for (int i = 0; i < 2; ++i) afr[i] = load_frag_A(&As[cur][0], wm + i * 16);
#pragma unroll
    for (int j = 0; j < 4; ++j) bfr[j] = load_frag_B(&Bs[cur][0], wn + j * 16);
#pragma unroll
    for (int i = 0; i < 2; ++i)
#pragma unroll
      for (int j = 0; j < 4; ++j)
        acc[i][j] = __builtin_amdgcn_wmma_f32_16x16x32_bf16(
            false, afr[i], false, bfr[j], (short)0, acc[i][j], false, false);
  }

  // Epilogue. C/D layout: col = lane&15, row(r) = r + 8*(lane>>4), r in 0..7.
  const int nlane = lane & 15;
  const int mhalf = (lane >> 4) * 8;
#pragma unroll
  for (int i = 0; i < 2; ++i) {
#pragma unroll
    for (int j = 0; j < 4; ++j) {
#pragma unroll
      for (int r = 0; r < 8; ++r) {
        const int row = bm + wm + i * 16 + mhalf + r;
        const int col = bn + wn + j * 16 + nlane;
        float v = acc[i][j][r] + (bias ? bias[col] : 0.f);
        if (gelu) {
          v = 0.5f * v * (1.f + erff(v * 0.70710678118654752f));
        } else {
          v = alpha * v + (R ? beta * R[(size_t)row * N + col] : 0.f);
        }
        const size_t o = (size_t)row * N + col;
        if (Ybf) Ybf[o] = (__bf16)v;
        if (Yf)  Yf[o]  = v;
      }
    }
  }
}

// ---------------------------------------------------------------------------
// f32 -> bf16 conversion
// ---------------------------------------------------------------------------
__global__ void cvt_bf16_kernel(const float* __restrict__ src, __bf16* __restrict__ dst, int n) {
  const int i = blockIdx.x * blockDim.x + threadIdx.x;
  if (i < n) dst[i] = (__bf16)src[i];
}

// ---------------------------------------------------------------------------
// Build the 256x256 circulant matrix equivalent to irfft(diag(exp_r) rfft(.)):
// M[n,m] = (1/256)[ e^{a0} cos b0 + (-1)^{n-m} e^{a128} cos b128
//                   + 2*sum_{k=1}^{127} e^{ak} cos(2*pi*k*(n-m)/256 + bk) ]
// ---------------------------------------------------------------------------
__global__ void build_M_kernel(const float* __restrict__ er, const float* __restrict__ ei,
                               __bf16* __restrict__ Mbf) {
  __shared__ float mag[129], ph[129];
  for (int k = threadIdx.x; k < 129; k += blockDim.x) {
    mag[k] = expf(er[k]);
    ph[k]  = ei[k];
  }
  __syncthreads();
  const int idx = blockIdx.x * blockDim.x + threadIdx.x;
  if (idx >= DDIM * DDIM) return;
  const int n = idx >> 8, m = idx & 255;
  const int d = n - m;
  float acc = mag[0] * cosf(ph[0]);
  acc += ((d & 1) ? -1.f : 1.f) * mag[128] * cosf(ph[128]);
  const float w = 6.28318530717958647f * (float)d / 256.f;
  for (int k = 1; k < 128; ++k) acc += 2.f * mag[k] * cosf(w * (float)k + ph[k]);
  Mbf[idx] = (__bf16)(acc * (1.f / 256.f));
}

// ---------------------------------------------------------------------------
// ctrl[b,n] = dt * sum_j ut[b,j] * B_ctrl[n,j]
// ---------------------------------------------------------------------------
__global__ void ctrl_kernel(const float* __restrict__ ut, const float* __restrict__ Bc,
                            const float* __restrict__ dt, float* __restrict__ out) {
  const int idx = blockIdx.x * blockDim.x + threadIdx.x;
  if (idx >= BATCH * DDIM) return;
  const int b = idx >> 8, n = idx & 255;
  float s = 0.f;
#pragma unroll
  for (int j = 0; j < UDIM; ++j) s += ut[b * UDIM + j] * Bc[n * UDIM + j];
  out[idx] = s * dt[0];
}

// ---------------------------------------------------------------------------
// yt[b,o] = sum_k znext[b,k]*C[o,k] + dt * sum_j ut[b,j]*Dm[o,j]
// ---------------------------------------------------------------------------
__global__ void yt_kernel(const float* __restrict__ znext, const float* __restrict__ ut,
                          const float* __restrict__ C, const float* __restrict__ Dm,
                          const float* __restrict__ dt, float* __restrict__ yt) {
  const int idx = blockIdx.x * blockDim.x + threadIdx.x;
  if (idx >= BATCH * NOBS) return;
  const int b = idx / NOBS, o = idx % NOBS;
  float s = 0.f;
  for (int k = 0; k < DDIM; ++k) s += znext[b * DDIM + k] * C[o * DDIM + k];
  float t = 0.f;
#pragma unroll
  for (int j = 0; j < UDIM; ++j) t += ut[b * UDIM + j] * Dm[o * UDIM + j];
  yt[idx] = s + t * dt[0];
}

// ---------------------------------------------------------------------------
// rev_residual += mean((a-b)^2)
// ---------------------------------------------------------------------------
__global__ void residual_kernel(const float* __restrict__ a, const float* __restrict__ b,
                                float* __restrict__ out, int n) {
  __shared__ float red[256];
  float s = 0.f;
  const float inv = 1.f / (float)n;
  for (int i = blockIdx.x * blockDim.x + threadIdx.x; i < n; i += gridDim.x * blockDim.x) {
    const float d = a[i] - b[i];
    s += d * d * inv;
  }
  red[threadIdx.x] = s;
  __syncthreads();
  for (int off = 128; off > 0; off >>= 1) {
    if (threadIdx.x < off) red[threadIdx.x] += red[threadIdx.x + off];
    __syncthreads();
  }
  if (threadIdx.x == 0) atomicAdd(out, red[0]);
}

// ---------------------------------------------------------------------------
// Host orchestration
// ---------------------------------------------------------------------------
extern "C" void kernel_launch(void* const* d_in, const int* in_sizes, int n_in,
                              void* d_out, int out_size, void* d_ws, size_t ws_size,
                              hipStream_t stream) {
  (void)in_sizes; (void)n_in; (void)out_size; (void)ws_size;

  const float* z_dyn = (const float*)d_in[0];
  const float* dt    = (const float*)d_in[2];   // scalar (device)
  const float* ut    = (const float*)d_in[3];
  const float* W1    = (const float*)d_in[4];
  const float* b1    = (const float*)d_in[5];
  const float* W2    = (const float*)d_in[6];
  const float* b2    = (const float*)d_in[7];
  const float* W3    = (const float*)d_in[8];
  const float* b3    = (const float*)d_in[9];
  const float* err   = (const float*)d_in[10];
  const float* eri   = (const float*)d_in[11];
  const float* Bc    = (const float*)d_in[12];
  const float* Cm    = (const float*)d_in[13];
  const float* Dm    = (const float*)d_in[14];

  // Workspace carve-up (all 256B aligned).
  char* ws = (char*)d_ws;
  size_t cur = 0;
  auto alloc = [&](size_t bytes) {
    void* p = ws + cur;
    cur += (bytes + 255) & ~(size_t)255;
    return p;
  };
  __bf16* W1bf    = (__bf16*)alloc((size_t)HID * DDIM * 2);
  __bf16* W2bf    = (__bf16*)alloc((size_t)HID * HID * 2);
  __bf16* W3bf    = (__bf16*)alloc((size_t)DDIM * HID * 2);
  __bf16* Mbf     = (__bf16*)alloc((size_t)DDIM * DDIM * 2);
  __bf16* zdyn_bf = (__bf16*)alloc((size_t)BATCH * DDIM * 2);
  __bf16* h1      = (__bf16*)alloc((size_t)BATCH * HID * 2);
  __bf16* h2      = (__bf16*)alloc((size_t)BATCH * HID * 2);
  float*  zlift_f = (float*) alloc((size_t)BATCH * DDIM * 4);
  __bf16* zlift_bf= (__bf16*)alloc((size_t)BATCH * DDIM * 2);
  float*  zev_f   = (float*) alloc((size_t)BATCH * DDIM * 4);  // reused as rev_check
  __bf16* zev_bf  = (__bf16*)alloc((size_t)BATCH * DDIM * 2);
  float*  zA_f    = (float*) alloc((size_t)BATCH * DDIM * 4);  // iterate (main + rev)
  __bf16* zA_bf   = (__bf16*)alloc((size_t)BATCH * DDIM * 2);
  float*  ctrl    = (float*) alloc((size_t)BATCH * DDIM * 4);

  float* out_znext = (float*)d_out;                    // [B, 256]
  float* out_yt    = out_znext + (size_t)BATCH * DDIM; // [B, 20]
  float* out_res   = out_yt + (size_t)BATCH * NOBS;    // [1]

  auto cvt = [&](const float* s, __bf16* d, int n) {
    cvt_bf16_kernel<<<(n + 255) / 256, 256, 0, stream>>>(s, d, n);
  };
  auto gemm = [&](const __bf16* X, const __bf16* W, const float* bias, const float* R,
                  float alpha, float beta, int gelu, __bf16* Ybf, float* Yf,
                  int M, int N, int K) {
    dim3 g(M / TILE_M, N / TILE_N);
    gemm_bf16_wmma<<<g, 256, 0, stream>>>(X, W, bias, R, alpha, beta, gelu, Ybf, Yf, M, N, K);
  };
  // MLP(x) fused: last layer computes  alpha*(h2@W3^T + b3) + beta*R  -> (outbf, outf)
  auto mlp = [&](const __bf16* xbf, const float* R, float alpha, float beta,
                 __bf16* outbf, float* outf) {
    gemm(xbf, W1bf, b1, nullptr, 1.f, 0.f, 1, h1, nullptr, BATCH, HID, DDIM);
    gemm(h1,  W2bf, b2, nullptr, 1.f, 0.f, 1, h2, nullptr, BATCH, HID, HID);
    gemm(h2,  W3bf, b3, R, alpha, beta, 0, outbf, outf, BATCH, DDIM, HID);
  };

  // 0) Convert weights and z_dyn to bf16; build circulant spectral matrix; ctrl term.
  cvt(W1, W1bf, HID * DDIM);
  cvt(W2, W2bf, HID * HID);
  cvt(W3, W3bf, DDIM * HID);
  cvt(z_dyn, zdyn_bf, BATCH * DDIM);
  build_M_kernel<<<(DDIM * DDIM + 255) / 256, 256, 0, stream>>>(err, eri, Mbf);
  ctrl_kernel<<<(BATCH * DDIM + 255) / 256, 256, 0, stream>>>(ut, Bc, dt, ctrl);

  // 1) lift: z_lifted = z_dyn + MLP(z_dyn)
  mlp(zdyn_bf, z_dyn, 1.f, 1.f, zlift_bf, zlift_f);

  // 2) spectral evolve + control: z_ev = z_lifted @ M^T + ctrl
  gemm(zlift_bf, Mbf, nullptr, ctrl, 1.f, 1.f, 0, zev_bf, zev_f, BATCH, DDIM, DDIM);

  // 3) inverse lift: z = z_ev; 5x: z = z_ev - MLP(z).  Last step writes d_out.
  mlp(zev_bf, zev_f, -1.f, 1.f, zA_bf, zA_f);
  mlp(zA_bf,  zev_f, -1.f, 1.f, zA_bf, zA_f);
  mlp(zA_bf,  zev_f, -1.f, 1.f, zA_bf, zA_f);
  mlp(zA_bf,  zev_f, -1.f, 1.f, zA_bf, zA_f);
  mlp(zA_bf,  zev_f, -1.f, 1.f, zA_bf, out_znext);

  // 4) reversibility branch: zr = inv_lift(z_lifted) (5 steps), rev_check = lift(zr)
  mlp(zlift_bf, zlift_f, -1.f, 1.f, zA_bf, zA_f);
  mlp(zA_bf,    zlift_f, -1.f, 1.f, zA_bf, zA_f);
  mlp(zA_bf,    zlift_f, -1.f, 1.f, zA_bf, zA_f);
  mlp(zA_bf,    zlift_f, -1.f, 1.f, zA_bf, zA_f);
  mlp(zA_bf,    zlift_f, -1.f, 1.f, zA_bf, zA_f);
  mlp(zA_bf,    zA_f,     1.f, 1.f, nullptr, zev_f);  // rev_check into zev_f (dead)

  // 5) rev_residual = mean((z_lifted - rev_check)^2)
  hipMemsetAsync(out_res, 0, sizeof(float), stream);
  residual_kernel<<<1024, 256, 0, stream>>>(zlift_f, zev_f, out_res, BATCH * DDIM);

  // 6) observations
  yt_kernel<<<(BATCH * NOBS + 255) / 256, 256, 0, stream>>>(out_znext, ut, Cm, Dm, dt, out_yt);
}
